// QuantLinear_32684701122970
// MI455X (gfx1250) — matrix operand order repaired and checked by reading, not compiled, not run on hardware
//
#include <hip/hip_runtime.h>
#include <hip/hip_bf16.h>

typedef __attribute__((ext_vector_type(16))) _Float16 v16h;
typedef __attribute__((ext_vector_type(8)))  _Float16 v8h;
typedef __attribute__((ext_vector_type(8)))  float    v8f;
typedef __attribute__((ext_vector_type(4)))  float    v4f;

#define GROUP 32

// ---------------- Prolog 1: fake-quant + dequant weight -> f16 [N,K] -------
// dq = (clip(rint(w/s + zp), 0, 15) - zp) * s ; s = 2^e so dq is EXACT in f16.
__global__ __launch_bounds__(256)
void dequant_w_f16(const float* __restrict__ w,
                   const float* __restrict__ scale,
                   const float* __restrict__ zp,
                   _Float16* __restrict__ wq,
                   int K, int n_groups) {
    long t    = (long)blockIdx.x * blockDim.x + threadIdx.x;
    long base = t * 8;                       // 8 elems, all inside one group
    int  row  = (int)(base / K);
    int  col  = (int)(base % K);
    int  g    = col / GROUP;
    float s   = scale[(long)row * n_groups + g];
    float z   = zp   [(long)row * n_groups + g];
    float inv = 1.0f / s;                    // power of two -> exact
    v4f w0 = *(const v4f*)(w + base);
    v4f w1 = *(const v4f*)(w + base + 4);
    v8h o;
    #pragma unroll
    for (int e = 0; e < 8; ++e) {
        float wv = (e < 4) ? w0[e] : w1[e - 4];
        float q  = fminf(fmaxf(rintf(wv * inv + z), 0.0f), 15.0f);
        o[e] = (_Float16)((q - z) * s);
    }
    *(v8h*)(wq + base) = o;
}

// ---------------- Prolog 2: x f32 -> f16 [M,K] -----------------------------
__global__ __launch_bounds__(256)
void cvt_x_f16(const float* __restrict__ x, _Float16* __restrict__ xh) {
    long base = ((long)blockIdx.x * blockDim.x + threadIdx.x) * 8;
    v4f x0 = *(const v4f*)(x + base);
    v4f x1 = *(const v4f*)(x + base + 4);
    v8h o;
    #pragma unroll
    for (int e = 0; e < 8; ++e)
        o[e] = (_Float16)((e < 4) ? x0[e] : x1[e - 4]);
    *(v8h*)(xh + base) = o;
}

// ---------------- fragment loaders (match ISA 16-bit A / B lane layouts) ---
// A 16x32: lanes 0-15 row m, halves[0..7]=K0..7, [8..15]=K16..23
//          lanes 16-31 row m, halves[0..7]=K8..15,[8..15]=K24..31
__device__ __forceinline__ v16h load_a_frag(const _Float16* p) {
    union { v16h v; v8h h[2]; } u;
    u.h[0] = *(const v8h*)(p);        // K + {0..7}   (lo lanes) / {8..15}
    u.h[1] = *(const v8h*)(p + 16);   // K + {16..23} (lo lanes) / {24..31}
    return u.v;
}
// B 32x16: lane l holds column l&15, halves[0..7]=K0..7, [8..15]=K16..23 (K += 8 for hi lanes)
__device__ __forceinline__ v16h load_b_frag(const _Float16* p) {
    union { v16h v; v8h h[2]; } u;
    u.h[0] = *(const v8h*)(p);
    u.h[1] = *(const v8h*)(p + 16);
    return u.v;
}

// ---------------- main WMMA GEMM:  out[M,N] = Xh[M,K] * Wq[N,K]^T + bias ---
// WG = 256 thr = 8 waves, tile 128N x 256M. Wave (wn,wm): wn in 0..1 (64 N),
// wm in 0..3 (64 M) -> 4 A-frags x 4 B-frags = 16 accumulators per wave.
// 16 b128 loads per 16 WMMAs per K-step of 32 (~33 FLOP / cache byte).
__global__ __launch_bounds__(256)
void wmma_gemm_f16(const _Float16* __restrict__ Wq,   // [N,K]  (A operand)
                   const _Float16* __restrict__ Xh,   // [M,K]  (B operand)
                   const float*    __restrict__ bias, // [N]
                   float*          __restrict__ out,  // [M,N]
                   int M, int N, int K) {
    const int lane = threadIdx.x & 31;
    const int wave = threadIdx.x >> 5;
    const int wn   = wave & 1;                     // 0..1 -> 64 N each
    const int wm   = wave >> 1;                    // 0..3 -> 64 M each
    const int nb   = blockIdx.x * 128 + wn * 64;   // wave N base (4 frags)
    const int mb   = blockIdx.y * 256 + wm * 64;   // wave M base (4 frags)
    const int l15  = lane & 15;
    const int lhi  = lane >> 4;

    // A row pointers: row = nb + i*16 + l15 ; half-lane K swizzle = lhi*8
    const _Float16* ap[4];
    ap[0] = Wq + (size_t)(nb + l15) * K + (lhi << 3);
    ap[1] = ap[0] + (size_t)16 * K;
    ap[2] = ap[0] + (size_t)32 * K;
    ap[3] = ap[0] + (size_t)48 * K;
    // B pointers: col = mb + j*16 + l15 ; half-lane K offset = lhi*8
    const _Float16* bp[4];
    bp[0] = Xh + (size_t)(mb + l15) * K + (lhi << 3);
    bp[1] = bp[0] + (size_t)16 * K;
    bp[2] = bp[0] + (size_t)32 * K;
    bp[3] = bp[0] + (size_t)48 * K;

    v8f c[4][4] = {};

    #pragma unroll 2
    for (int k0 = 0; k0 < K; k0 += 32) {
        v16h a[4], b[4];
        #pragma unroll
        for (int i = 0; i < 4; ++i) a[i] = load_a_frag(ap[i] + k0);
        #pragma unroll
        for (int j = 0; j < 4; ++j) b[j] = load_b_frag(bp[j] + k0);

        #pragma unroll
        for (int i = 0; i < 4; ++i) {
            #pragma unroll
            for (int j = 0; j < 4; ++j) {
                c[i][j] = __builtin_amdgcn_wmma_f32_16x16x32_f16(
                    false, a[i], false, b[j], (short)0, c[i][j], false, false);
            }
        }
    }

    // Epilogue. D layout: VGPR r, lane l -> n = nb + i*16 + 8*lhi + r (consecutive
    // over r), m = mb + j*16 + l15. Bias add + 2x nontemporal b128 stores
    // (out is 128 MB written once; keep Wq/Xh resident in L2 instead).
    #pragma unroll
    for (int i = 0; i < 4; ++i) {
        int n0 = nb + i * 16 + (lhi << 3);
        v4f bv0 = *(const v4f*)(bias + n0);
        v4f bv1 = *(const v4f*)(bias + n0 + 4);
        #pragma unroll
        for (int j = 0; j < 4; ++j) {
            size_t m = (size_t)(mb + j * 16 + l15);
            float* o = out + m * (size_t)N + n0;
            v4f lo, hi;
            #pragma unroll
            for (int r = 0; r < 4; ++r) {
                lo[r] = c[i][j][r]     + bv0[r];
                hi[r] = c[i][j][r + 4] + bv1[r];
            }
            __builtin_nontemporal_store(lo, (v4f*)(o));
            __builtin_nontemporal_store(hi, (v4f*)(o + 4));
        }
    }
}

// ---------------------------------------------------------------------------
extern "C" void kernel_launch(void* const* d_in, const int* in_sizes, int n_in,
                              void* d_out, int out_size, void* d_ws, size_t ws_size,
                              hipStream_t stream) {
    const float* x     = (const float*)d_in[0];   // [B,S,IN]
    const float* w     = (const float*)d_in[1];   // [OUT,IN]
    const float* bias  = (const float*)d_in[2];   // [OUT]
    const float* scale = (const float*)d_in[3];   // [OUT, n_groups]
    const float* zp    = (const float*)d_in[4];   // [OUT, n_groups]

    const int  OUT_CH  = in_sizes[2];
    const long wElems  = (long)in_sizes[1];
    const int  IN_CH   = (int)(wElems / OUT_CH);
    const long xElems  = (long)in_sizes[0];
    const int  M       = (int)(xElems / IN_CH);
    const int  n_groups = IN_CH / GROUP;

    _Float16* wq = (_Float16*)d_ws;                       // [OUT_CH, IN_CH] f16
    _Float16* xh = wq + (size_t)OUT_CH * IN_CH;           // [M, IN_CH]      f16

    // Prolog kernels (8 elems/thread, shapes divide evenly: 4096x4096, 8192x4096)
    dequant_w_f16<<<(int)(wElems / (256L * 8)), 256, 0, stream>>>(
        w, scale, zp, wq, IN_CH, n_groups);
    cvt_x_f16<<<(int)(xElems / (256L * 8)), 256, 0, stream>>>(x, xh);

    dim3 grid(OUT_CH / 128, M / 256);  // (32, 32)
    wmma_gemm_f16<<<grid, 256, 0, stream>>>(wq, xh, bias, (float*)d_out,
                                            M, OUT_CH, IN_CH);
}
